// FFTAppearanceEncoder_11785390260848
// MI455X (gfx1250) — matrix operand order repaired and checked by reading, not compiled, not run on hardware
//
#include <hip/hip_runtime.h>

// ---------------------------------------------------------------------------
// FFTAppearanceEncoder for MI455X (gfx1250, wave32, WMMA), NHWC f16 layout.
//
// Input flattening assumption (setup_inputs() declaration order):
//   d_in[0] frames [32,3,128,128] f32
//   d_in[1] boxes  [4,8,8,5]      f32
//   d_in[2] drone_mask [4,8,8]    f32
//   d_in[3 + 6*i + j]  pix layer i (i=0..4), field j in (w,b,g,beta,m,v)
//   d_in[33 + 6*i + j] fft layer i (i=0..2), same field order
//   d_in[51] fuse_w [256,768], d_in[52] fuse_b, d_in[53] ln_g, d_in[54] ln_b
//
// Conv = implicit GEMM, K ordered tap-major (k = tap*Cin + ci) so each
// 32-wide K block shares one (dy,dx) tap: A fragment = two b128 loads from
// NHWC activations, B fragment = two b128 loads from fragment-packed weights.
// Hot kernels are templated on compile-time shapes so all tile decode is
// shifts/masks and the K loop is fully unrolled straight-line WMMA.
// ---------------------------------------------------------------------------

typedef __attribute__((ext_vector_type(16))) _Float16 v16h;
typedef __attribute__((ext_vector_type(8)))  _Float16 v8h;
typedef __attribute__((ext_vector_type(8)))  float    v8f;

#define NIMG 32
#define PI_F 3.14159265358979323846f

// --------------------------- DFT tables ------------------------------------
__global__ void k_dft_init(float* __restrict__ Ct, float* __restrict__ St) {
  int idx = blockIdx.x * blockDim.x + threadIdx.x;   // 16384
  int u = idx >> 7, j = idx & 127;
  int p = (u * j) & 127;
  float th = 2.0f * PI_F * (float)p * (1.0f / 128.0f);
  float sc = 0.088388347648318447f;                  // 1/sqrt(128)
  Ct[idx] = __cosf(th) * sc;
  St[idx] = __sinf(th) * sc;
}

// frames NCHW f32 -> NHWC f16 ([n][y][x][c], c=0..2)
__global__ void k_frames_nhwc(const float* __restrict__ in, _Float16* __restrict__ out) {
  int idx = blockIdx.x * blockDim.x + threadIdx.x;   // 32*16384*3
  if (idx >= NIMG * 16384 * 3) return;
  int c = idx % 3;
  int p = idx / 3;
  int pix = p & 16383, n = p >> 14;
  out[idx] = (_Float16)in[((long)n * 3 + c) * 16384 + pix];
}

// ------------------------ weight packing -----------------------------------
// Generic layer (Cin % 32 == 0): fragment-order pack.
// wq[(kb_idx*Cout + co)*32 + within] = w_fold[k = kb_idx*32 + within][co],
// with k = tap*Cin + ci. Also folds BN into (s,t).
__global__ void k_pack_w_frag(const float* __restrict__ w, const float* __restrict__ bb,
                              const float* __restrict__ g, const float* __restrict__ beta,
                              const float* __restrict__ m, const float* __restrict__ vv,
                              _Float16* __restrict__ wq, float* __restrict__ s,
                              float* __restrict__ t, int cin, int cout) {
  long idx = (long)blockIdx.x * blockDim.x + threadIdx.x;
  if (idx < (long)cout) {
    float sc = g[idx] * rsqrtf(vv[idx] + 1e-5f);
    s[idx] = sc;
    t[idx] = (bb[idx] - m[idx]) * sc + beta[idx];
  }
  long total = (long)9 * cin * cout;
  if (idx >= total) return;
  int  within = (int)(idx & 31);
  long t2 = idx >> 5;
  int  co = (int)(t2 % cout);
  long kb = t2 / cout;
  long k  = (kb << 5) + within;            // k = tap*cin + ci
  int  tap = (int)(k / cin);
  int  ci  = (int)(k - (long)tap * cin);
  wq[idx] = (_Float16)w[((long)co * cin + ci) * 9 + tap];
}

// First layer (Cin=3, ktot=27, kpad=32): wq0[co*32 + k], k = tap*3 + ci.
__global__ void k_pack_w0(const float* __restrict__ w, const float* __restrict__ bb,
                          const float* __restrict__ g, const float* __restrict__ beta,
                          const float* __restrict__ m, const float* __restrict__ vv,
                          _Float16* __restrict__ wq, float* __restrict__ s,
                          float* __restrict__ t, int cout) {
  int idx = blockIdx.x * blockDim.x + threadIdx.x;
  if (idx < cout) {
    float sc = g[idx] * rsqrtf(vv[idx] + 1e-5f);
    s[idx] = sc;
    t[idx] = (bb[idx] - m[idx]) * sc + beta[idx];
  }
  if (idx >= cout * 32) return;
  int k = idx & 31, co = idx >> 5;
  _Float16 val = (_Float16)0.f;
  if (k < 27) {
    int tap = k / 3, ci = k - tap * 3;
    val = (_Float16)w[((long)co * 3 + ci) * 9 + tap];
  }
  wq[idx] = val;
}

// fuse_w [256][768] f32 -> fragment order: wqf[(kb*256+co)*32 + within]
__global__ void k_pack_fuse_frag(const float* __restrict__ fw, _Float16* __restrict__ wqf) {
  int idx = blockIdx.x * blockDim.x + threadIdx.x;   // 768*256
  if (idx >= 768 * 256) return;
  int within = idx & 31;
  int t2 = idx >> 5;
  int co = t2 & 255, kb = t2 >> 8;
  wqf[idx] = (_Float16)fw[co * 768 + kb * 32 + within];
}

// --------------------------- FFT stream ------------------------------------
__global__ void k_fft_rows(const float* __restrict__ x, const float* __restrict__ Ct,
                           const float* __restrict__ St, float* __restrict__ Yr,
                           float* __restrict__ Yi) {
  long idx = (long)blockIdx.x * blockDim.x + threadIdx.x;  // 96*128*128
  int u = (int)(idx & 127);
  long r = idx >> 7;
  int y = (int)(r & 127);
  long nc = r >> 7;
  const float* row = x + (nc << 14) + ((long)y << 7);
  const float* cu = Ct + (u << 7);
  const float* su = St + (u << 7);
  float ar = 0.f, ai = 0.f;
  #pragma unroll 8
  for (int j = 0; j < 128; ++j) {
    float v = row[j];
    ar += v * cu[j];
    ai -= v * su[j];
  }
  Yr[idx] = ar;                                            // [nc][y][u]
  Yi[idx] = ai;
}

// column DFT + magnitude + fftshift + log1p -> NHWC f16 [n][y][x][3]
__global__ void k_fft_cols_mag(const float* __restrict__ Yr, const float* __restrict__ Yi,
                               const float* __restrict__ Ct, const float* __restrict__ St,
                               _Float16* __restrict__ fo) {
  long idx = (long)blockIdx.x * blockDim.x + threadIdx.x;  // 96*128*128
  int v = (int)(idx & 127);
  long r = idx >> 7;
  int u = (int)(r & 127);
  int nc = (int)(r >> 7);
  const float* cu = Ct + (u << 7);
  const float* su = St + (u << 7);
  const float* yr = Yr + ((long)nc << 14) + v;
  const float* yi = Yi + ((long)nc << 14) + v;
  float fr = 0.f, fi = 0.f;
  #pragma unroll 4
  for (int yy = 0; yy < 128; ++yy) {
    float a = yr[(long)yy << 7], b = yi[(long)yy << 7];
    fr += cu[yy] * a + su[yy] * b;
    fi += cu[yy] * b - su[yy] * a;
  }
  float mag = sqrtf(fr * fr + fi * fi);
  int us = (u + 64) & 127, vs = (v + 64) & 127;            // fftshift
  int n = nc / 3, c = nc - n * 3;
  fo[((long)n * 16384 + us * 128 + vs) * 3 + c] = (_Float16)log1pf(mag);
}

// -------------------- conv3x3 + BN + ReLU, WMMA, NHWC ----------------------
// Generic path: CIN % 32 == 0, HW power of two. One wave = 16 row-contiguous
// pixels x 16 couts; all shape math is compile-time shifts/masks; the 9*CIN/32
// K blocks are fully unrolled straight-line WMMA.
// A layout: M=lane%16; e<8 -> k=half*8+e, e>=8 -> k=16+half*8+e (contig ci)
// B layout: N=lane%16; k as for A (half*8 + e%8, +16 for e>=8) from fragment-packed wq
// C layout: M = r + 8*half, N = lane%16
template <int CIN, int COUT, int HW>
__global__ void k_conv_wmma(const _Float16* __restrict__ in, _Float16* __restrict__ out,
                            const _Float16* __restrict__ wq, const float* __restrict__ s,
                            const float* __restrict__ t) {
  constexpr int TILES_PIX = (HW * HW) >> 4;
  constexpr int TILES_CO  = COUT >> 4;
  constexpr int NBLK      = CIN >> 5;
  const int wave = threadIdx.x >> 5;
  const int lane = threadIdx.x & 31;
  const int half = lane >> 4;
  const int l16  = lane & 15;

  int tile = blockIdx.x * 8 + wave;
  int pix_t  = tile & (TILES_PIX - 1);
  int rem    = tile / TILES_PIX;               // power of two -> shift
  int co_blk = rem & (TILES_CO - 1);
  int n      = rem / TILES_CO;
  int y  = (pix_t * 16) / HW;
  int x0 = (pix_t * 16) & (HW - 1);
  int xx0 = x0 + l16;
  int co = co_blk * 16 + l16;

  const _Float16* inb = in + (long)n * HW * HW * CIN;
  const _Float16* bptr = wq + (long)co * 32 + (half << 3);
  constexpr long BSTEP = (long)COUT * 32;

  v8f acc = {};
  #pragma unroll
  for (int tap = 0; tap < 9; ++tap) {
    constexpr int d = 1;                        // pad = 1
    const int dy = tap / 3 - d, dx = tap % 3 - d;  // constants after unroll
    int yy = y + dy;
    int xc = xx0 + dx;
    bool ok = (yy >= 0) && (yy < HW) && (xc >= 0) && (xc < HW);
    const _Float16* prow = inb + ((long)yy * HW + xc) * CIN + (half << 3);
    #pragma unroll
    for (int cb = 0; cb < NBLK; ++cb) {
      v8h lo = {}, hi = {};
      if (ok) {
        lo = *(const v8h*)(prow);
        hi = *(const v8h*)(prow + 16);
      }
      prow += 32;
      v16h a;
      #pragma unroll
      for (int i = 0; i < 8; ++i) { a[i] = lo[i]; a[8 + i] = hi[i]; }
      v8h bl = *(const v8h*)(bptr);
      v8h bh = *(const v8h*)(bptr + 16);
      bptr += BSTEP;
      v16h b;
      #pragma unroll
      for (int i = 0; i < 8; ++i) { b[i] = bl[i]; b[8 + i] = bh[i]; }
      acc = __builtin_amdgcn_wmma_f32_16x16x32_f16(false, a, false, b, (short)0, acc,
                                                   false, false);
    }
  }
  float sc = s[co], tc = t[co];
  _Float16* ob = out + (((long)n * HW + y) * HW + x0) * COUT + co;
  #pragma unroll
  for (int r = 0; r < 8; ++r) {
    float v = acc[r] * sc + tc;
    ob[(long)(r + (half << 3)) * COUT] = (_Float16)(v > 0.f ? v : 0.f);
  }
}

// First layer: Cin=3 (ktot=27 <= 32), single WMMA per tile, scalar A gather.
template <int COUT, int HW>
__global__ void k_conv0_wmma(const _Float16* __restrict__ in, _Float16* __restrict__ out,
                             const _Float16* __restrict__ wq, const float* __restrict__ s,
                             const float* __restrict__ t) {
  constexpr int TILES_PIX = (HW * HW) >> 4;
  constexpr int TILES_CO  = COUT >> 4;
  const int wave = threadIdx.x >> 5;
  const int lane = threadIdx.x & 31;
  const int half = lane >> 4;
  const int l16  = lane & 15;

  int tile = blockIdx.x * 8 + wave;
  int pix_t  = tile & (TILES_PIX - 1);
  int rem    = tile / TILES_PIX;
  int co_blk = rem & (TILES_CO - 1);
  int n      = rem / TILES_CO;
  int y  = (pix_t * 16) / HW;
  int x0 = (pix_t * 16) & (HW - 1);
  int xx0 = x0 + l16;
  int co = co_blk * 16 + l16;

  const _Float16* inb = in + (long)n * HW * HW * 3;
  v16h a;
  #pragma unroll
  for (int e = 0; e < 16; ++e) {
    const int k = ((e >> 3) << 4) + (half << 3) + (e & 7);
    _Float16 av = (_Float16)0.f;
    if (k < 27) {                               // k/3, k%3 fold per-element
      int tap = k / 3, ci = k - tap * 3;
      int dy = tap / 3 - 1, dx = tap % 3 - 1;
      int yy = y + dy, xc = xx0 + dx;
      if (yy >= 0 && yy < HW && xc >= 0 && xc < HW)
        av = inb[((long)yy * HW + xc) * 3 + ci];
    }
    a[e] = av;
  }
  const _Float16* bp = wq + ((long)co << 5) + (half << 3);
  v8h bl = *(const v8h*)bp;
  v8h bh = *(const v8h*)(bp + 16);
  v16h b;
  #pragma unroll
  for (int i = 0; i < 8; ++i) { b[i] = bl[i]; b[8 + i] = bh[i]; }
  v8f acc = {};
  acc = __builtin_amdgcn_wmma_f32_16x16x32_f16(false, a, false, b, (short)0, acc,
                                               false, false);
  float sc = s[co], tc = t[co];
  _Float16* ob = out + (((long)n * HW + y) * HW + x0) * COUT + co;
  #pragma unroll
  for (int r = 0; r < 8; ++r) {
    float v = acc[r] * sc + tc;
    ob[(long)(r + (half << 3)) * COUT] = (_Float16)(v > 0.f ? v : 0.f);
  }
}

// 2x2 maxpool, NHWC (channel fastest -> coalesced); C, HIN compile-time.
template <int C, int HIN>
__global__ void k_maxpool(const _Float16* __restrict__ in, _Float16* __restrict__ out) {
  constexpr int HO = HIN >> 1;
  constexpr long TOTAL = (long)NIMG * C * HO * HO;
  long idx = (long)blockIdx.x * blockDim.x + threadIdx.x;
  if (idx >= TOTAL) return;
  int c = (int)(idx & (C - 1));
  long p = idx / C;
  int xo = (int)(p & (HO - 1));
  long q = p / HO;
  int yo = (int)(q & (HO - 1));
  long n = q / HO;
  const _Float16* b = in + (((long)n * HIN + yo * 2) * HIN + xo * 2) * C + c;
  constexpr long RS = (long)HIN * C;
  float v0 = (float)b[0], v1 = (float)b[C], v2 = (float)b[RS], v3 = (float)b[RS + C];
  out[idx] = (_Float16)fmaxf(fmaxf(v0, v1), fmaxf(v2, v3));
}

// ROI mean pool on NHWC 32x32x256 map -> cat[nm][col_off + c]
__global__ void k_roi(const _Float16* __restrict__ feat, const float* __restrict__ boxes,
                      const float* __restrict__ mask, _Float16* __restrict__ cat,
                      int col_off) {
  int nm = blockIdx.x;            // 0..255
  int c  = threadIdx.x;           // 0..255
  const float* bx = boxes + (long)nm * 5;
  float cx = bx[1] * 32.f, cy = bx[2] * 32.f;
  float bw = fmaxf(bx[3] * 32.f, 2.f), bh = fmaxf(bx[4] * 32.f, 2.f);
  int x1 = (int)(cx - bw * 0.5f); x1 = x1 < 0 ? 0 : (x1 > 31 ? 31 : x1);
  int y1 = (int)(cy - bh * 0.5f); y1 = y1 < 0 ? 0 : (y1 > 31 ? 31 : y1);
  int x2 = (int)(cx + bw * 0.5f); x2 = x2 < 1 ? 1 : (x2 > 32 ? 32 : x2);
  int y2 = (int)(cy + bh * 0.5f); y2 = y2 < 1 ? 1 : (y2 > 32 ? 32 : y2);
  if (x2 < x1 + 1) x2 = x1 + 1;
  if (y2 < y1 + 1) y2 = y1 + 1;
  const _Float16* f = feat + (long)(nm >> 3) * 32 * 32 * 256 + c;
  float sum = 0.f;
  for (int yy = y1; yy < y2; ++yy)
    for (int xx = x1; xx < x2; ++xx)
      sum += (float)f[(yy * 32 + xx) * 256];
  float area = (float)((y2 - y1) * (x2 - x1));
  float mb = (mask[nm] >= 0.5f) ? 1.f : 0.f;
  cat[(long)nm * 768 + col_off + c] = (_Float16)(sum / area * mb);
}

// Scene: global avg pool of pix_map (NHWC) -> cols 512..767 of all M rows
__global__ void k_scene(const _Float16* __restrict__ pix_map, _Float16* __restrict__ cat) {
  int idx = blockIdx.x * blockDim.x + threadIdx.x;   // 32*256
  int n = idx >> 8, c = idx & 255;
  const _Float16* f = pix_map + (long)n * 32 * 32 * 256 + c;
  float s = 0.f;
  for (int i = 0; i < 1024; ++i) s += (float)f[(long)i * 256];
  _Float16 h = (_Float16)(s * (1.f / 1024.f));
  for (int m = 0; m < 8; ++m)
    cat[(long)(n * 8 + m) * 768 + 512 + c] = h;
}

// ---------------------- fuse GEMM (256x256x768) ----------------------------
__global__ void k_fuse_wmma(const _Float16* __restrict__ cat, const _Float16* __restrict__ wqf,
                            const float* __restrict__ fb, float* __restrict__ yout) {
  const int wave = threadIdx.x >> 5;
  const int lane = threadIdx.x & 31;
  const int half = lane >> 4;
  const int l16  = lane & 15;
  int tile = blockIdx.x * 8 + wave;
  int row_blk = tile >> 4;
  int col_blk = tile & 15;
  int arow = row_blk * 16 + l16;
  int bcol = col_blk * 16 + l16;
  const _Float16* ap = cat + (long)arow * 768 + (half << 3);
  const _Float16* bp = wqf + (long)bcol * 32 + (half << 3);
  v8f acc = {};
  #pragma unroll 4
  for (int kb = 0; kb < 768; kb += 32) {
    v8h lo = *(const v8h*)(ap);
    v8h hi = *(const v8h*)(ap + 16);
    ap += 32;
    v16h a;
    #pragma unroll
    for (int i = 0; i < 8; ++i) { a[i] = lo[i]; a[8 + i] = hi[i]; }
    v8h bl = *(const v8h*)(bp);
    v8h bh = *(const v8h*)(bp + 16);
    bp += 256 * 32;
    v16h b;
    #pragma unroll
    for (int i = 0; i < 8; ++i) { b[i] = bl[i]; b[8 + i] = bh[i]; }
    acc = __builtin_amdgcn_wmma_f32_16x16x32_f16(false, a, false, b, (short)0, acc,
                                                 false, false);
  }
  float bias = fb[bcol];
  #pragma unroll
  for (int r = 0; r < 8; ++r) {
    int row = row_blk * 16 + r + (half << 3);
    yout[(long)row * 256 + bcol] = acc[r] + bias;
  }
}

// LayerNorm(256) + ReLU, one 256-thread block per row, LDS tree reduction.
__global__ void k_layernorm(const float* __restrict__ y, const float* __restrict__ g,
                            const float* __restrict__ b, float* __restrict__ out) {
  __shared__ float red[256];
  int row = blockIdx.x, c = threadIdx.x;
  float v = y[(long)row * 256 + c];
  red[c] = v;
  __syncthreads();
  for (int s2 = 128; s2 > 0; s2 >>= 1) {
    if (c < s2) red[c] += red[c + s2];
    __syncthreads();
  }
  float mu = red[0] * (1.f / 256.f);
  __syncthreads();
  float d = v - mu;
  red[c] = d * d;
  __syncthreads();
  for (int s2 = 128; s2 > 0; s2 >>= 1) {
    if (c < s2) red[c] += red[c + s2];
    __syncthreads();
  }
  float var = red[0] * (1.f / 256.f);
  float o = d * rsqrtf(var + 1e-5f) * g[c] + b[c];
  out[(long)row * 256 + c] = o > 0.f ? o : 0.f;
}

// ---------------------------------------------------------------------------
// Host launcher
// ---------------------------------------------------------------------------
extern "C" void kernel_launch(void* const* d_in, const int* in_sizes, int n_in,
                              void* d_out, int out_size, void* d_ws, size_t ws_size,
                              hipStream_t stream) {
  (void)in_sizes; (void)n_in; (void)out_size; (void)ws_size;
  const float* frames = (const float*)d_in[0];
  const float* boxes  = (const float*)d_in[1];
  const float* dmask  = (const float*)d_in[2];
  const float* fuse_w = (const float*)d_in[51];
  const float* fuse_b = (const float*)d_in[52];
  const float* ln_g   = (const float*)d_in[53];
  const float* ln_b   = (const float*)d_in[54];

  char* ws = (char*)d_ws;
  size_t off = 0;
  auto alloc = [&](size_t bytes) -> void* {
    void* p = ws + off;
    off += (bytes + 255) & ~(size_t)255;
    return p;
  };
  float*    Ct     = (float*)alloc(16384 * 4);
  float*    St     = (float*)alloc(16384 * 4);
  _Float16* xin    = (_Float16*)alloc((size_t)NIMG * 16384 * 3 * 2);   // NHWC
  _Float16* fftin  = (_Float16*)alloc((size_t)NIMG * 16384 * 3 * 2);   // NHWC
  float*    Yr     = (float*)alloc((size_t)96 * 16384 * 4);
  float*    Yi     = (float*)alloc((size_t)96 * 16384 * 4);
  _Float16* bufA   = (_Float16*)alloc((size_t)NIMG * 64 * 16384 * 2);  // max activation
  _Float16* bufB   = (_Float16*)alloc((size_t)NIMG * 64 * 16384 * 2);
  _Float16* pixmap = (_Float16*)alloc((size_t)NIMG * 1024 * 256 * 2);  // NHWC 32x32x256
  _Float16* fftmap = (_Float16*)alloc((size_t)NIMG * 1024 * 256 * 2);
  _Float16* cat    = (_Float16*)alloc((size_t)256 * 768 * 2);
  float*    yfc    = (float*)alloc((size_t)256 * 256 * 4);
  _Float16* wqf    = (_Float16*)alloc((size_t)768 * 256 * 2);

  // conv layer configs: {cin, cout, H(=W), d_in base index}
  struct LCfg { int cin, cout, hw, base; };
  const LCfg cfg[8] = {{3, 64, 128, 3},   {64, 64, 128, 9},  {64, 128, 64, 15},
                       {128, 128, 64, 21}, {128, 256, 32, 27},
                       {3, 64, 128, 33},  {64, 128, 64, 39}, {128, 256, 32, 45}};
  _Float16* wq[8];
  float *sv[8], *tv[8];
  for (int i = 0; i < 8; ++i) {
    const LCfg& L = cfg[i];
    size_t nelem = (L.cin == 3) ? (size_t)32 * L.cout : (size_t)9 * L.cin * L.cout;
    wq[i] = (_Float16*)alloc(nelem * 2);
    sv[i] = (float*)alloc((size_t)L.cout * 4);
    tv[i] = (float*)alloc((size_t)L.cout * 4);
  }

  // --- setup ---
  k_dft_init<<<64, 256, 0, stream>>>(Ct, St);
  k_frames_nhwc<<<6144, 256, 0, stream>>>(frames, xin);
  for (int i = 0; i < 8; ++i) {
    const LCfg& L = cfg[i];
    if (L.cin == 3) {
      k_pack_w0<<<(L.cout * 32 + 255) / 256, 256, 0, stream>>>(
          (const float*)d_in[L.base + 0], (const float*)d_in[L.base + 1],
          (const float*)d_in[L.base + 2], (const float*)d_in[L.base + 3],
          (const float*)d_in[L.base + 4], (const float*)d_in[L.base + 5],
          wq[i], sv[i], tv[i], L.cout);
    } else {
      long tot = (long)9 * L.cin * L.cout;
      k_pack_w_frag<<<(int)((tot + 255) / 256), 256, 0, stream>>>(
          (const float*)d_in[L.base + 0], (const float*)d_in[L.base + 1],
          (const float*)d_in[L.base + 2], (const float*)d_in[L.base + 3],
          (const float*)d_in[L.base + 4], (const float*)d_in[L.base + 5],
          wq[i], sv[i], tv[i], L.cin, L.cout);
    }
  }
  k_pack_fuse_frag<<<768, 256, 0, stream>>>(fuse_w, wqf);

  // --- FFT stream input ---
  k_fft_rows<<<6144, 256, 0, stream>>>(frames, Ct, St, Yr, Yi);
  k_fft_cols_mag<<<6144, 256, 0, stream>>>(Yr, Yi, Ct, St, fftin);

  // grid = wave-tiles / 8 (exact; keeps EXEC all-ones for every WMMA wave)
  auto blocks = [](int cout, int hw) {
    return (int)(((long)NIMG * ((hw * hw) >> 4) * (cout >> 4)) / 8);
  };

  // --- pixel stream ---
  k_conv0_wmma<64, 128><<<blocks(64, 128), 256, 0, stream>>>(xin, bufA, wq[0], sv[0], tv[0]);
  k_conv_wmma<64, 64, 128><<<blocks(64, 128), 256, 0, stream>>>(bufA, bufB, wq[1], sv[1], tv[1]);
  k_maxpool<64, 128><<<(NIMG * 64 * 64 * 64) / 256, 256, 0, stream>>>(bufB, bufA);
  k_conv_wmma<64, 128, 64><<<blocks(128, 64), 256, 0, stream>>>(bufA, bufB, wq[2], sv[2], tv[2]);
  k_conv_wmma<128, 128, 64><<<blocks(128, 64), 256, 0, stream>>>(bufB, bufA, wq[3], sv[3], tv[3]);
  k_maxpool<128, 64><<<(NIMG * 128 * 32 * 32) / 256, 256, 0, stream>>>(bufA, bufB);
  k_conv_wmma<128, 256, 32><<<blocks(256, 32), 256, 0, stream>>>(bufB, pixmap, wq[4], sv[4], tv[4]);

  // --- FFT stream ---
  k_conv0_wmma<64, 128><<<blocks(64, 128), 256, 0, stream>>>(fftin, bufA, wq[5], sv[5], tv[5]);
  k_maxpool<64, 128><<<(NIMG * 64 * 64 * 64) / 256, 256, 0, stream>>>(bufA, bufB);
  k_conv_wmma<64, 128, 64><<<blocks(128, 64), 256, 0, stream>>>(bufB, bufA, wq[6], sv[6], tv[6]);
  k_maxpool<128, 64><<<(NIMG * 128 * 32 * 32) / 256, 256, 0, stream>>>(bufA, bufB);
  k_conv_wmma<128, 256, 32><<<blocks(256, 32), 256, 0, stream>>>(bufB, fftmap, wq[7], sv[7], tv[7]);

  // --- pooling heads -> cat [256 x 768] ---
  k_roi<<<256, 256, 0, stream>>>(pixmap, boxes, dmask, cat, 0);
  k_roi<<<256, 256, 0, stream>>>(fftmap, boxes, dmask, cat, 256);
  k_scene<<<32, 256, 0, stream>>>(pixmap, cat);

  // --- fuse GEMM + LayerNorm + ReLU ---
  k_fuse_wmma<<<32, 256, 0, stream>>>(cat, wqf, fuse_b, yfc);
  k_layernorm<<<256, 256, 0, stream>>>(yfc, ln_g, ln_b, (float*)d_out);
}
